// channel_net_3624952398060
// MI455X (gfx1250) — compile-verified, matrix operand order checked by reading, not run on hardware
//
#include <hip/hip_runtime.h>

typedef _Float16 v16h __attribute__((ext_vector_type(16)));
typedef _Float16 v8h  __attribute__((ext_vector_type(8)));
typedef _Float16 v4h  __attribute__((ext_vector_type(4)));
typedef float    v8f  __attribute__((ext_vector_type(8)));

#define B_TOTAL 131072
#define M_DIM   512
#define H_ENC   128
#define N_DIM   32
#define H_DEC   128
#define TB      128     // rows per workgroup
#define NTHREADS 256    // 8 wave32

// LDS byte offsets (dynamic shared memory, 168 KB total).
// GEMM1 ping-pong region [0,64K) is reused by GEMM4's W4 ping-pong buffers.
//   As(i)  = 0     + i*32768   (x chunk,  128x64 f16, 16 KB each)
//   Ws(i)  = 16384 + i*32768   (W1 chunk, 128x64 f16, 16 KB each)
//   W4b(i) = 0     + i*32768   (W4 chunk, 128x128 f16, 32 KB each; aliases above)
#define OFF_Z1   65536u    // Z1: 128 x 128 f16 (32 KB)
#define OFF_W2   98304u    // W2:  32 x 128 f16 ( 8 KB)
#define OFF_W3   106496u   // W3: 128 x  32 f16 ( 8 KB)
#define OFF_Z2   114688u   // Z2: 128 x  32 f32 (16 KB)
#define OFF_Y    131072u   // Y:  128 x  32 f16 ( 8 KB)
#define OFF_D    139264u   // D:  128 x 128 f16 (32 KB)
#define SMEM_BYTES 172032u

// Load a 16x32 f16 WMMA A/B fragment from LDS (row-major, K contiguous).
// ISA layout: lanes 0-15 hold K{0..7,16..23}, lanes 16-31 hold K{8..15,24..31}
// => two 16-byte LDS loads (ds_load_b128) per lane.
static __device__ __forceinline__ v16h load_frag16(const _Float16* p, int laneSel) {
  v8h lo = *(const v8h*)(p + laneSel);
  v8h hi = *(const v8h*)(p + laneSel + 16);
  return __builtin_shufflevector(lo, hi, 0,1,2,3,4,5,6,7,8,9,10,11,12,13,14,15);
}

static __device__ __forceinline__ v8f vzero8() {
  v8f z;
#pragma unroll
  for (int i = 0; i < 8; ++i) z[i] = 0.0f;
  return z;
}

static __device__ __forceinline__ void cvt_store4(_Float16* dst, const float4 f) {
  v4h h;
  h[0] = (_Float16)f.x; h[1] = (_Float16)f.y;
  h[2] = (_Float16)f.z; h[3] = (_Float16)f.w;
  *(v4h*)dst = h;
}

static __device__ __forceinline__ v8f wmma_f16(v16h a, v16h b, v8f c) {
  return __builtin_amdgcn_wmma_f32_16x16x32_f16(false, a, false, b, (short)0, c,
                                                false, false);
}

// Stage a 128x64 f32 tile from global (row stride srcStride) into LDS as f16.
static __device__ __forceinline__ void stage64(_Float16* dst, const float* src,
                                               int srcStride, int kOff, int tid) {
  for (int v = tid; v < (128 * 64) / 4; v += NTHREADS) {
    int r = v >> 4, c = (v & 15) << 2;
    cvt_store4(dst + r * 64 + c,
               *(const float4*)(src + (size_t)r * srcStride + kOff + c));
  }
}

// Stage a 128x128 f32 tile from global (row stride 128) into LDS as f16.
static __device__ __forceinline__ void stage128(_Float16* dst, const float* src,
                                                int tid) {
  for (int v = tid; v < (128 * 128) / 4; v += NTHREADS) {
    int r = v >> 5, c = (v & 31) << 2;
    cvt_store4(dst + r * 128 + c, *(const float4*)(src + (size_t)r * 128 + c));
  }
}

__global__ void __launch_bounds__(NTHREADS)
channel_net_fused_kernel(const float* __restrict__ x,
                         const float* __restrict__ noise,
                         const float* __restrict__ hvec,
                         const float* __restrict__ W1, const float* __restrict__ b1,
                         const float* __restrict__ W2, const float* __restrict__ b2,
                         const float* __restrict__ W3, const float* __restrict__ b3,
                         const float* __restrict__ W4, const float* __restrict__ b4,
                         float* __restrict__ out) {
  extern __shared__ char smem[];
  _Float16* Z1  = (_Float16*)(smem + OFF_Z1);
  _Float16* W2s = (_Float16*)(smem + OFF_W2);
  _Float16* W3s = (_Float16*)(smem + OFF_W3);
  float*    Z2  = (float*)   (smem + OFF_Z2);
  _Float16* Ys  = (_Float16*)(smem + OFF_Y);
  _Float16* Ds  = (_Float16*)(smem + OFF_D);

  const int tid     = threadIdx.x;
  const int wave    = tid >> 5;
  const int lane    = tid & 31;
  const int rowBase = wave * 16;            // wave's 16-row stripe within tile
  const int row0    = blockIdx.x * TB;      // global base row
  const int laneSel = (lane >> 4) << 3;     // 0 or 8 halves (fragment K select)
  const int cRow    = (lane >> 4) << 3;     // C/D layout row offset base
  const int cCol    = lane & 15;            // C/D layout column within tile
  const int col16   = lane & 15;

  const float* xBlk = x + (size_t)row0 * M_DIM;

  // ---- stage small weights W2 [32x128], W3 [128x32] into LDS as f16 ----
  for (int v = tid; v < (N_DIM * H_ENC) / 4; v += NTHREADS) {
    int r = v >> 5, c = (v & 31) << 2;
    cvt_store4(W2s + r * H_ENC + c, *(const float4*)(W2 + r * H_ENC + c));
  }
  for (int v = tid; v < (H_DEC * N_DIM) / 4; v += NTHREADS) {
    int r = v >> 3, c = (v & 7) << 2;
    cvt_store4(W3s + r * N_DIM + c, *(const float4*)(W3 + r * N_DIM + c));
  }

  // =================== GEMM1: Z1 = relu(x @ W1^T + b1), K=512 ===================
  v8f acc1[8];
#pragma unroll
  for (int t = 0; t < 8; ++t) acc1[t] = vzero8();

  // prologue: stage chunk 0 into buffer 0 (runtime address math, no pointer arrays)
  stage64((_Float16*)(smem + 0u), xBlk, M_DIM, 0, tid);
  stage64((_Float16*)(smem + 16384u), W1, M_DIM, 0, tid);
  __syncthreads();

  for (int kc = 0; kc < 8; ++kc) {      // 8 chunks of K=64, ping-pong staged
    const unsigned cur = (unsigned)(kc & 1), nxt = cur ^ 1u;
    if (kc + 1 < 8) {                   // stage next chunk while computing this one
      const int kOff = (kc + 1) * 64;
      stage64((_Float16*)(smem + nxt * 32768u), xBlk, M_DIM, kOff, tid);
      stage64((_Float16*)(smem + 16384u + nxt * 32768u), W1, M_DIM, kOff, tid);
      if (kc + 2 < 8)                   // L2 prefetch two chunks ahead
        __builtin_prefetch(xBlk + (size_t)(tid >> 1) * M_DIM + (kc + 2) * 64 +
                               ((tid & 1) << 5), 0, 1);
    }
    const _Float16* Ab = (const _Float16*)(smem + cur * 32768u);
    const _Float16* Wb = (const _Float16*)(smem + 16384u + cur * 32768u);
#pragma unroll
    for (int kk = 0; kk < 64; kk += 32) {
      v16h a = load_frag16(Ab + (rowBase + col16) * 64 + kk, laneSel);
#pragma unroll
      for (int half = 0; half < 2; ++half) {
        v16h bfr[4];                    // grouped B loads -> partial dscnt waits
#pragma unroll
        for (int t = 0; t < 4; ++t)
          bfr[t] = load_frag16(Wb + ((half * 4 + t) * 16 + col16) * 64 + kk, laneSel);
#pragma unroll
        for (int t = 0; t < 4; ++t)
          acc1[half * 4 + t] = wmma_f16(a, bfr[t], acc1[half * 4 + t]);
      }
    }
    __syncthreads();                    // single barrier per chunk
  }

  // bias + relu, write Z1 (f16)
#pragma unroll
  for (int t = 0; t < 8; ++t) {
#pragma unroll
    for (int g = 0; g < 8; ++g) {
      int r = rowBase + cRow + g;
      int c = t * 16 + cCol;
      float v = acc1[t][g] + b1[c];
      Z1[r * H_ENC + c] = (_Float16)(v > 0.0f ? v : 0.0f);
    }
  }
  __syncthreads();

  // =================== GEMM2: Z2 = Z1 @ W2^T + b2, K=128 ===================
  v8f acc2[2];
#pragma unroll
  for (int t = 0; t < 2; ++t) acc2[t] = vzero8();
#pragma unroll
  for (int k0 = 0; k0 < 128; k0 += 32) {
    v16h a = load_frag16(Z1 + (rowBase + col16) * H_ENC + k0, laneSel);
    v16h b0 = load_frag16(W2s + (0 * 16 + col16) * H_ENC + k0, laneSel);
    v16h b1f = load_frag16(W2s + (1 * 16 + col16) * H_ENC + k0, laneSel);
    acc2[0] = wmma_f16(a, b0, acc2[0]);
    acc2[1] = wmma_f16(a, b1f, acc2[1]);
  }
#pragma unroll
  for (int t = 0; t < 2; ++t) {
#pragma unroll
    for (int g = 0; g < 8; ++g) {
      int r = rowBase + cRow + g;
      int c = t * 16 + cCol;
      Z2[r * N_DIM + c] = acc2[t][g] + b2[c];
    }
  }
  __syncthreads();

  // ============ power-normalize, single-tap channel rotate, AWGN ============
  if (tid < TB) {
    const int r = tid;
    float zv[N_DIM];
    float s = 0.0f;
#pragma unroll
    for (int j = 0; j < N_DIM; ++j) { zv[j] = Z2[r * N_DIM + j]; s += zv[j] * zv[j]; }
    const float scale = 4.0f * rsqrtf(s);       // sqrt(N*0.5) = 4
    const float h0 = hvec[0], h1 = hvec[1];
    const float* nz = noise + (size_t)(row0 + r) * N_DIM;
#pragma unroll
    for (int j = 0; j < N_DIM / 2; ++j) {
      float re = zv[2 * j] * scale, im = zv[2 * j + 1] * scale;
      Ys[r * N_DIM + 2 * j]     = (_Float16)(re * h0 - im * h1 + nz[2 * j]);
      Ys[r * N_DIM + 2 * j + 1] = (_Float16)(re * h1 + im * h0 + nz[2 * j + 1]);
    }
  }
  __syncthreads();

  // =================== GEMM3: D = relu(Y @ W3^T + b3), K=32 ===================
  v8f acc3[8];
#pragma unroll
  for (int t = 0; t < 8; ++t) acc3[t] = vzero8();
  {
    v16h a = load_frag16(Ys + (rowBase + col16) * N_DIM, laneSel);
#pragma unroll
    for (int half = 0; half < 2; ++half) {
      v16h bfr[4];
#pragma unroll
      for (int t = 0; t < 4; ++t)
        bfr[t] = load_frag16(W3s + ((half * 4 + t) * 16 + col16) * N_DIM, laneSel);
#pragma unroll
      for (int t = 0; t < 4; ++t)
        acc3[half * 4 + t] = wmma_f16(a, bfr[t], acc3[half * 4 + t]);
    }
  }
#pragma unroll
  for (int t = 0; t < 8; ++t) {
#pragma unroll
    for (int g = 0; g < 8; ++g) {
      int r = rowBase + cRow + g;
      int c = t * 16 + cCol;
      float v = acc3[t][g] + b3[c];
      Ds[r * H_DEC + c] = (_Float16)(v > 0.0f ? v : 0.0f);
    }
  }
  __syncthreads();

  // =================== GEMM4: out = D @ W4^T + b4, K=128 ===================
  // 4 column chunks of 128 outputs, ping-pong staged W4 tiles
  // (aliases the dead GEMM1 staging region: buf(i) = smem + i*32768).
  stage128((_Float16*)(smem + 0u), W4, tid);   // prologue: chunk 0
  __syncthreads();

  for (int nc = 0; nc < 4; ++nc) {
    const unsigned cur = (unsigned)(nc & 1), nxt = cur ^ 1u;
    if (nc + 1 < 4) {
      stage128((_Float16*)(smem + nxt * 32768u),
               W4 + (size_t)(nc + 1) * 128 * H_DEC, tid);
      if (nc + 2 < 4)
        __builtin_prefetch(W4 + (size_t)((nc + 2) * 128 + (tid >> 1)) * H_DEC +
                               ((tid & 1) << 6), 0, 1);
    }
    const _Float16* Wb = (const _Float16*)(smem + cur * 32768u);
    v8f acc4[8];
#pragma unroll
    for (int t = 0; t < 8; ++t) acc4[t] = vzero8();
#pragma unroll
    for (int k0 = 0; k0 < 128; k0 += 32) {
      v16h a = load_frag16(Ds + (rowBase + col16) * H_DEC + k0, laneSel);
#pragma unroll
      for (int half = 0; half < 2; ++half) {
        v16h bfr[4];
#pragma unroll
        for (int t = 0; t < 4; ++t)
          bfr[t] = load_frag16(Wb + ((half * 4 + t) * 16 + col16) * 128 + k0, laneSel);
#pragma unroll
        for (int t = 0; t < 4; ++t)
          acc4[half * 4 + t] = wmma_f16(a, bfr[t], acc4[half * 4 + t]);
      }
    }
#pragma unroll
    for (int t = 0; t < 8; ++t) {
#pragma unroll
      for (int g = 0; g < 8; ++g) {
        int r = rowBase + cRow + g;
        int c = nc * 128 + t * 16 + cCol;
        out[(size_t)(row0 + r) * M_DIM + c] = acc4[t][g] + b4[c];
      }
    }
    __syncthreads();
  }
}

extern "C" void kernel_launch(void* const* d_in, const int* in_sizes, int n_in,
                              void* d_out, int out_size, void* d_ws, size_t ws_size,
                              hipStream_t stream) {
  (void)in_sizes; (void)n_in; (void)d_ws; (void)ws_size; (void)out_size;
  const float* x     = (const float*)d_in[0];
  const float* noise = (const float*)d_in[1];
  const float* hvec  = (const float*)d_in[2];
  const float* W1    = (const float*)d_in[3];
  const float* b1    = (const float*)d_in[4];
  const float* W2    = (const float*)d_in[5];
  const float* b2    = (const float*)d_in[6];
  const float* W3    = (const float*)d_in[7];
  const float* b3    = (const float*)d_in[8];
  const float* W4    = (const float*)d_in[9];
  const float* b4    = (const float*)d_in[10];
  float* out = (float*)d_out;

  (void)hipFuncSetAttribute((const void*)channel_net_fused_kernel,
                            hipFuncAttributeMaxDynamicSharedMemorySize,
                            (int)SMEM_BYTES);

  dim3 grid(B_TOTAL / TB);   // 1024 workgroups
  dim3 block(NTHREADS);      // 8 wave32
  channel_net_fused_kernel<<<grid, block, SMEM_BYTES, stream>>>(
      x, noise, hvec, W1, b1, W2, b2, W3, b3, W4, b4, out);
}
